// PointNetSetAbstractionMsg_75969381532366
// MI455X (gfx1250) — matrix operand order repaired and checked by reading, not compiled, not run on hardware
//
#include <hip/hip_runtime.h>
#include <hip/hip_bf16.h>

typedef __attribute__((ext_vector_type(16))) _Float16 v16h;
typedef __attribute__((ext_vector_type(8)))  float    v8f;

#define BB 8
#define NN 4096
#define SS 1024
#define SFD 32
#define DD 64

// ---------------------------------------------------------------------------
// 1) Farthest point sampling: one block per batch, running-min dists in LDS.
// ---------------------------------------------------------------------------
__global__ void fps_kernel(const float* __restrict__ xyz, int* __restrict__ fps) {
  const int b = blockIdx.x, tid = threadIdx.x;
  extern __shared__ char smem[];
  float* dist = (float*)smem;              // 4096
  float* redv = dist + NN;                 // 256
  int*   redi = (int*)(redv + 256);        // 256
  int*   sfar = (int*)(redi + 256);        // 1
  const float* px = xyz + (size_t)b * 3 * NN;
  const float* py = px + NN;
  const float* pz = py + NN;
  for (int n = tid; n < NN; n += 256) dist[n] = 1e10f;
  if (tid == 0) *sfar = 0;
  __syncthreads();
  for (int step = 0; step < SS; ++step) {
    const int far = *sfar;
    if (tid == 0) fps[b * SS + step] = far;   // scan emits incoming index
    const float cx = px[far], cy = py[far], cz = pz[far];
    float bv = -1.f; int bi = 0;
    for (int n = tid; n < NN; n += 256) {
      float dx = px[n] - cx, dy = py[n] - cy, dz = pz[n] - cz;
      float d = dx * dx + dy * dy + dz * dz;
      float dd = dist[n];
      if (d < dd) dd = d;
      dist[n] = dd;
      if (dd > bv) { bv = dd; bi = n; }
    }
    redv[tid] = bv; redi[tid] = bi;
    __syncthreads();
    for (int off = 128; off > 0; off >>= 1) {
      if (tid < off) {
        float v2 = redv[tid + off]; int i2 = redi[tid + off];
        if (v2 > redv[tid] || (v2 == redv[tid] && i2 < redi[tid])) {
          redv[tid] = v2; redi[tid] = i2;
        }
      }
      __syncthreads();
    }
    if (tid == 0) *sfar = redi[0];
    __syncthreads();
  }
}

// ---------------------------------------------------------------------------
// 2) Gather new_xyz (-> out1 transposed + ws copy) and new_bin (-> out3).
// ---------------------------------------------------------------------------
__global__ void gather_kernel(const float* __restrict__ xyz,
                              const float* __restrict__ feature,
                              const int* __restrict__ fps,
                              float* __restrict__ out1, float* __restrict__ out3,
                              float* __restrict__ nx) {
  const int t = blockIdx.x * 256 + threadIdx.x;
  if (t >= BB * SS) return;
  const int b = t >> 10, s = t & (SS - 1);
  const int idx = fps[t];
  for (int j = 0; j < 3; ++j) {
    float v = xyz[((size_t)b * 3 + j) * NN + idx];
    out1[((size_t)b * 3 + j) * SS + s] = v;
    nx[(size_t)t * 3 + j] = v;
  }
  const float* f = feature + ((size_t)b * NN + idx) * SFD;
  float* o = out3 + (size_t)t * SFD;
  for (int c = 0; c < SFD; ++c) o[c] = f[c];
}

// ---------------------------------------------------------------------------
// 3) Prep: fold BN + bias into scale/shift; swizzle f16 weights into the exact
//    per-lane WMMA B-fragment layout (lane n = l&15, K pair = 2(v&3) + 16(v>>2) + 8*(l>=16)).
// ---------------------------------------------------------------------------
struct PrepArgs {
  const float* W[6];  const float* bs[6]; const float* gm[6];
  const float* bt[6]; const float* mu[6]; const float* var[6];
};

__global__ void prep_kernel(PrepArgs pa, unsigned* __restrict__ swz,
                            float* __restrict__ scsh) {
  const int L = blockIdx.x, tid = threadIdx.x;
  const int cinS[6]   = {67, 64, 64, 67, 128, 128};
  const int cpad[6]   = {96, 64, 64, 96, 128, 128};
  const int cO[6]     = {64, 64, 128, 128, 128, 256};
  const int swzoff[6] = {0, 3072, 5120, 9216, 15360, 23552};
  const int ssoff[6]  = {0, 128, 256, 512, 768, 1024};
  const int cin = cinS[L], Co = cO[L], Nt = Co >> 4;
  const int npair = (cpad[L] * Co) >> 1;
  const float* W = pa.W[L];
  for (int i = tid; i < npair; i += 256) {
    const int v = i & 7, lane = (i >> 3) & 31, rest = i >> 8;
    const int nt = rest % Nt, kc = rest / Nt;
    const int k = kc * 32 + 2 * (v & 3) + ((v >> 2) << 4) + (lane >> 4) * 8;
    const int o = nt * 16 + (lane & 15);
    const float w0 = (k     < cin) ? W[o * cin + k]     : 0.f;
    const float w1 = (k + 1 < cin) ? W[o * cin + k + 1] : 0.f;
    union { _Float16 h[2]; unsigned u; } p;
    p.h[0] = (_Float16)w0; p.h[1] = (_Float16)w1;
    swz[swzoff[L] + i] = p.u;
  }
  if (tid < Co) {
    const float sc = pa.gm[L][tid] * rsqrtf(pa.var[L][tid] + 1e-5f);
    const float sh = sc * (pa.bs[L][tid] - pa.mu[L][tid]) + pa.bt[L][tid];
    scsh[ssoff[L] + tid]      = sc;
    scsh[ssoff[L] + Co + tid] = sh;
  }
}

// ---------------------------------------------------------------------------
// 4) Fused distance + 64-NN selection per (b,s) row.
// ---------------------------------------------------------------------------
__global__ void knn_kernel(const float* __restrict__ xyz,
                           const float* __restrict__ feature,
                           const float* __restrict__ nbin,
                           const float* __restrict__ nx,
                           int* __restrict__ knn) {
  const int bs = blockIdx.x, tid = threadIdx.x;
  const int b = bs >> 10;
  extern __shared__ char smem[];
  float* dist = (float*)smem;            // 4096
  float* redv = dist + NN;               // 256
  int*   redi = (int*)(redv + 256);      // 256
  float* q    = (float*)(redi + 256);    // 35
  if (tid < SFD) q[tid] = nbin[(size_t)bs * SFD + tid];
  if (tid < 3)   q[SFD + tid] = nx[(size_t)bs * 3 + tid];
  __syncthreads();
  const float* fb = feature + (size_t)b * NN * SFD;
  const float* px = xyz + (size_t)b * 3 * NN;
  for (int n = tid; n < NN; n += 256) {
    const float* f = fb + (size_t)n * SFD;
    float d = 0.f;
#pragma unroll
    for (int c = 0; c < SFD; ++c) { float t2 = q[c] - f[c]; d += t2 * t2; }
#pragma unroll
    for (int j = 0; j < 3; ++j) { float t2 = q[SFD + j] - px[j * NN + n]; d += t2 * t2; }
    dist[n] = d;
  }
  __syncthreads();
  for (int pass = 0; pass < 64; ++pass) {
    float bv = 3.4e38f; int bi = 0x7fffffff;
    for (int n = tid; n < NN; n += 256) {
      float v = dist[n];
      if (v < bv) { bv = v; bi = n; }
    }
    redv[tid] = bv; redi[tid] = bi;
    __syncthreads();
    for (int off = 128; off > 0; off >>= 1) {
      if (tid < off) {
        float v2 = redv[tid + off]; int i2 = redi[tid + off];
        if (v2 < redv[tid] || (v2 == redv[tid] && i2 < redi[tid])) {
          redv[tid] = v2; redi[tid] = i2;
        }
      }
      __syncthreads();
    }
    if (tid == 0) { int j = redi[0]; knn[(size_t)bs * 64 + pass] = j; dist[j] = 3.3e38f; }
    __syncthreads();
  }
}

// ---------------------------------------------------------------------------
// 5) WMMA GEMM layer: D = A(KxCin) * W^T, epilogue scale/shift + ReLU -> f16.
//    A fragment follows documented 16-bit A layout; B comes pre-swizzled.
// ---------------------------------------------------------------------------
__device__ __forceinline__ void gemm_bn_relu(
    const _Float16* __restrict__ src, int lds_src,
    _Float16* __restrict__ dst, int lds_dst,
    const unsigned* __restrict__ wsw,
    const float* __restrict__ scale, const float* __restrict__ shift,
    int M, int Cin_pad, int Cout, int tid) {
  const int lane = tid & 31;
  const int wave = tid >> 5;
  const int Mt = M >> 4, Nt = Cout >> 4, Kc = Cin_pad >> 5;
  const int halfsel = lane >> 4;
  const int lid = lane & 15;
  for (int t = wave; t < Mt * Nt; t += 8) {
    const int mt = t / Nt, nt = t - mt * Nt;
    v8f acc = {};
    const _Float16* arow = src + (mt * 16 + lid) * lds_src + halfsel * 8;
    const unsigned* wp = wsw + (nt * 32 + lane) * 8;
    for (int kc = 0; kc < Kc; ++kc) {
      union { v16h h; unsigned u[8]; } A, Bm;
      const _Float16* ap = arow + kc * 32;
#pragma unroll
      for (int v = 0; v < 8; ++v) {
        const int c = ((v & 3) << 1) + ((v >> 2) << 4);
        A.u[v] = *(const unsigned*)(ap + c);
      }
      const unsigned* bp = wp + kc * (Nt << 8);
#pragma unroll
      for (int v = 0; v < 8; ++v) Bm.u[v] = bp[v];
      acc = __builtin_amdgcn_wmma_f32_16x16x32_f16(false, A.h, false, Bm.h,
                                                   (short)0, acc, false, false);
    }
    const int n = nt * 16 + lid;
    const float sc = scale[n], sh = shift[n];
#pragma unroll
    for (int r = 0; r < 8; ++r) {
      const int m = mt * 16 + r + halfsel * 8;
      float y = acc[r] * sc + sh;
      dst[m * lds_dst + n] = (_Float16)(y > 0.f ? y : 0.f);
    }
  }
}

// ---------------------------------------------------------------------------
// 6) Per-(b,s) grouped MLP: gather K neighbors, 3 WMMA layers, max over K.
// ---------------------------------------------------------------------------
__global__ void mlp_kernel(const float* __restrict__ points, const float* __restrict__ xyz,
                           const float* __restrict__ nx, const int* __restrict__ knn,
                           const unsigned* __restrict__ swz0, const unsigned* __restrict__ swz1,
                           const unsigned* __restrict__ swz2,
                           const float* __restrict__ ss0, const float* __restrict__ ss1,
                           const float* __restrict__ ss2,
                           float* __restrict__ out2,
                           int KS, int C0, int C1, int C2, int chan_off) {
  const int tid = threadIdx.x;
  const int bs = blockIdx.x;
  const int b = bs >> 10, s = bs & (SS - 1);
  extern __shared__ char smem[];
  _Float16* buf0 = (_Float16*)smem;           // KS x 136 (holds 96 then C1)
  _Float16* buf1 = buf0 + KS * 136;           // KS x 272 (holds C0 then C2)
  int*   sidx = (int*)(buf1 + KS * 272);
  float* sctr = (float*)(sidx + KS);
  if (tid < KS) sidx[tid] = knn[(size_t)bs * 64 + tid];
  if (tid < 3)  sctr[tid] = nx[(size_t)bs * 3 + tid];
  __syncthreads();
  // gather: channels 0..63 point features, 64..66 relative xyz, 67..95 zero pad
  for (int e = tid; e < KS * 96; e += 256) {
    const int m = e / 96, c = e - m * 96;
    const int n = sidx[m];
    float val;
    if (c < 64)      val = points[((size_t)b * DD + c) * NN + n];
    else if (c < 67) val = xyz[((size_t)b * 3 + (c - 64)) * NN + n] - sctr[c - 64];
    else             val = 0.f;
    buf0[m * 136 + c] = (_Float16)val;
  }
  __syncthreads();
  gemm_bn_relu(buf0, 136, buf1, 272, swz0, ss0, ss0 + C0, KS, 96, C0, tid);
  __syncthreads();
  gemm_bn_relu(buf1, 272, buf0, 136, swz1, ss1, ss1 + C1, KS, C0, C1, tid);
  __syncthreads();
  gemm_bn_relu(buf0, 136, buf1, 272, swz2, ss2, ss2 + C2, KS, C1, C2, tid);
  __syncthreads();
  for (int c = tid; c < C2; c += 256) {
    float mx = 0.f;  // post-ReLU values are >= 0
    for (int m = 0; m < KS; ++m) {
      float v = (float)buf1[m * 272 + c];
      mx = fmaxf(mx, v);
    }
    out2[((size_t)b * 384 + chan_off + c) * SS + s] = mx;
  }
}

// ---------------------------------------------------------------------------
extern "C" void kernel_launch(void* const* d_in, const int* in_sizes, int n_in,
                              void* d_out, int out_size, void* d_ws, size_t ws_size,
                              hipStream_t stream) {
  const float* xyz     = (const float*)d_in[0];
  const float* feature = (const float*)d_in[1];
  const float* points  = (const float*)d_in[2];

  char* wsb = (char*)d_ws;
  int*      fps  = (int*)wsb;                                   // 8192 ints
  int*      knn  = (int*)(wsb + 32768);                         // 8192*64 ints
  float*    nx   = (float*)(wsb + 32768 + 2097152);             // 8192*3 floats
  unsigned* swz  = (unsigned*)(wsb + 32768 + 2097152 + 98304);  // 39936 uints
  float*    scsh = (float*)(wsb + 32768 + 2097152 + 98304 + 159744); // 1536 floats

  float* out  = (float*)d_out;
  float* out1 = out;                        // (B,3,S)
  float* out2 = out + BB * 3 * SS;          // (B,384,S)
  float* out3 = out2 + BB * 384 * SS;       // (B,S,32)

  PrepArgs pa;
  for (int L = 0; L < 6; ++L) {
    const int base = 3 + L * 6;
    pa.W[L]   = (const float*)d_in[base + 0];
    pa.bs[L]  = (const float*)d_in[base + 1];
    pa.gm[L]  = (const float*)d_in[base + 2];
    pa.bt[L]  = (const float*)d_in[base + 3];
    pa.mu[L]  = (const float*)d_in[base + 4];
    pa.var[L] = (const float*)d_in[base + 5];
  }

  fps_kernel<<<BB, 256, 18464, stream>>>(xyz, fps);
  gather_kernel<<<(BB * SS + 255) / 256, 256, 0, stream>>>(xyz, feature, fps, out1, out3, nx);
  prep_kernel<<<6, 256, 0, stream>>>(pa, swz, scsh);
  knn_kernel<<<BB * SS, 256, 18576, stream>>>(xyz, feature, out3, nx, knn);
  mlp_kernel<<<BB * SS, 256, 32 * 820 + 16, stream>>>(
      points, xyz, nx, knn, swz + 0, swz + 3072, swz + 5120,
      scsh + 0, scsh + 128, scsh + 256, out2, 32, 64, 64, 128, 0);
  mlp_kernel<<<BB * SS, 256, 64 * 820 + 16, stream>>>(
      points, xyz, nx, knn, swz + 9216, swz + 15360, swz + 23552,
      scsh + 512, scsh + 768, scsh + 1024, out2, 64, 128, 128, 256, 128);
}
